// SKITNO_56873956933788
// MI455X (gfx1250) — matrix operand order repaired and checked by reading, not compiled, not run on hardware
//
#include <hip/hip_runtime.h>
#include <hip/hip_bf16.h>
#include <math.h>
#include <stdint.h>

// ---------------------------------------------------------------------------
// Problem constants (match reference)
// ---------------------------------------------------------------------------
#define BSZ     2
#define SEQ     2048
#define DMODEL  1024
#define D1      2048
#define NHEAD   16
#define HD      128
#define NKI     16
#define RANKK   32
#define LN_GAMMA (-0.0010005003335835335f)   // ln(0.999)
#define M_ROWS  (BSZ * SEQ)                  // 4096

typedef float  v8f   __attribute__((ext_vector_type(8)));
typedef __bf16 v16bf __attribute__((ext_vector_type(16)));

// ---------------------------------------------------------------------------
// CDNA5 async global->LDS copy (ASYNCcnt path), 16B per lane.
// ---------------------------------------------------------------------------
__device__ __forceinline__ void async_copy_b128(void* lds_ptr, const void* gptr)
{
    uint32_t lds_off = (uint32_t)(uintptr_t)lds_ptr;   // addr[31:0] = LDS offset
    uint64_t gaddr   = (uint64_t)(uintptr_t)gptr;
    asm volatile("global_load_async_to_lds_b128 %0, %1, off"
                 :: "v"(lds_off), "v"(gaddr) : "memory");
}

// ---------------------------------------------------------------------------
// Elementwise fp32 -> (bf16 hi, bf16 lo) split, same layout.
// ---------------------------------------------------------------------------
__global__ __launch_bounds__(256)
void split_f32_bf16(const float* __restrict__ src, __bf16* __restrict__ hi,
                    __bf16* __restrict__ lo, size_t n)
{
    size_t stride = (size_t)gridDim.x * blockDim.x;
    for (size_t i = (size_t)blockIdx.x * blockDim.x + threadIdx.x; i < n; i += stride) {
        float  x = src[i];
        __bf16 h = (__bf16)x;
        hi[i] = h;
        lo[i] = (__bf16)(x - (float)h);
    }
}

// ---------------------------------------------------------------------------
// W [K,Nc] fp32 -> transposed split bf16: th/tl [Nc,K].
// ---------------------------------------------------------------------------
__global__ __launch_bounds__(256)
void split_transpose_w(const float* __restrict__ W, __bf16* __restrict__ th,
                       __bf16* __restrict__ tl, int K, int Nc)
{
    size_t total  = (size_t)K * Nc;
    size_t stride = (size_t)gridDim.x * blockDim.x;
    for (size_t i = (size_t)blockIdx.x * blockDim.x + threadIdx.x; i < total; i += stride) {
        int    n = (int)(i / K);
        int    k = (int)(i % K);
        float  x = W[(size_t)k * Nc + n];
        __bf16 h = (__bf16)x;
        th[i] = h;
        tl[i] = (__bf16)(x - (float)h);
    }
}

// ---------------------------------------------------------------------------
// Split-bf16 GEMM: C[M,Nc] = act( X @ W + bias )
//   X@W ~= Xh@Wh + Xh@Wl + Xl@Wh  (fp32 accumulate, lo*lo dropped ~2^-16).
// WMMA f32_16x16x32_bf16, block 256 thr (8 waves), tile 128x64, K-chunk 32.
// Double-buffered async pipeline: issue tile k+1, s_wait_asynccnt <= 6
// (per-thread in-flight ops of the new tile; async loads retire in order,
// so this guarantees tile k has landed), barrier, compute tile k.
// ---------------------------------------------------------------------------
#define MBLK 128
#define NBLK 64
#define KBLK 32

__global__ __launch_bounds__(256)
void gemm_bf16_split(const __bf16* __restrict__ Xh, const __bf16* __restrict__ Xl,
                     const __bf16* __restrict__ Wth, const __bf16* __restrict__ Wtl,
                     const float* __restrict__ bias, float* __restrict__ C,
                     int M, int K, int Nc, int doSilu)
{
    __shared__ __bf16 XhS[2][MBLK * KBLK];   // 2 x 8 KB
    __shared__ __bf16 XlS[2][MBLK * KBLK];   // 2 x 8 KB
    __shared__ __bf16 WhS[2][NBLK * KBLK];   // 2 x 4 KB (row = n, 32 K contiguous)
    __shared__ __bf16 WlS[2][NBLK * KBLK];   // 2 x 4 KB   -> 48 KB total

    const int m0   = blockIdx.y * MBLK;
    const int n0   = blockIdx.x * NBLK;
    const int t    = threadIdx.x;
    const int wave = t >> 5;
    const int lane = t & 31;
    const int mrow = lane & 15;
    const int sel  = lane >> 4;

    // 6 async b128 ops per thread per K-chunk (4 for X planes, 2 for W planes)
    auto issue_tile = [&](int buf, int k0) {
        #pragma unroll
        for (int i = 0; i < 2; ++i) {
            int q  = t + i * 256;
            int r  = q >> 2;
            int c8 = (q & 3) * 8;
            size_t g = (size_t)(m0 + r) * K + k0 + c8;
            async_copy_b128(&XhS[buf][q * 8], Xh + g);
            async_copy_b128(&XlS[buf][q * 8], Xl + g);
        }
        int q  = t;
        int r  = q >> 2;
        int c8 = (q & 3) * 8;
        size_t g = (size_t)(n0 + r) * K + k0 + c8;
        async_copy_b128(&WhS[buf][q * 8], Wth + g);
        async_copy_b128(&WlS[buf][q * 8], Wtl + g);
    };

    v8f acc[NBLK / 16] = {};

    const int nk = K / KBLK;
    issue_tile(0, 0);

    #pragma unroll 1
    for (int kc = 0; kc < nk; ++kc) {
        const int cur = kc & 1;
        if (kc + 1 < nk) {
            issue_tile(cur ^ 1, (kc + 1) * KBLK);
            asm volatile("s_wait_asynccnt 0x6" ::: "memory");  // tile kc landed
        } else {
            asm volatile("s_wait_asynccnt 0x0" ::: "memory");
        }
        __syncthreads();

        // A fragments (16-bit A 16x32):
        //  lane<16: K {0..7, 16..23};  lane>=16: K {8..15, 24..31}
        const __bf16* xh = &XhS[cur][(wave * 16 + mrow) * KBLK];
        const __bf16* xl = &XlS[cur][(wave * 16 + mrow) * KBLK];
        v16bf ah, al;
        #pragma unroll
        for (int i = 0; i < 8; ++i) {
            ah[i]     = xh[sel * 8 + i];
            ah[8 + i] = xh[16 + sel * 8 + i];
            al[i]     = xl[sel * 8 + i];
            al[8 + i] = xl[16 + sel * 8 + i];
        }
        // B fragments (16-bit B 32x16): lane col n, same K order as A
        #pragma unroll
        for (int tn = 0; tn < NBLK / 16; ++tn) {
            const __bf16* wh = &WhS[cur][(tn * 16 + mrow) * KBLK];
            const __bf16* wl = &WlS[cur][(tn * 16 + mrow) * KBLK];
            v16bf bh, bl;
            #pragma unroll
            for (int e = 0; e < 8; ++e) {
                bh[e]     = wh[sel * 8 + e];
                bh[8 + e] = wh[16 + sel * 8 + e];
                bl[e]     = wl[sel * 8 + e];
                bl[8 + e] = wl[16 + sel * 8 + e];
            }
            acc[tn] = __builtin_amdgcn_wmma_f32_16x16x32_bf16(
                false, ah, false, bh, (short)0, acc[tn], false, false);
            acc[tn] = __builtin_amdgcn_wmma_f32_16x16x32_bf16(
                false, ah, false, bl, (short)0, acc[tn], false, false);
            acc[tn] = __builtin_amdgcn_wmma_f32_16x16x32_bf16(
                false, al, false, bh, (short)0, acc[tn], false, false);
        }
        __syncthreads();   // all reads of 'cur' done before it is re-filled
    }

    // Epilogue: C/D layout — VGPR e: lanes 0-15 -> M=e, lanes 16-31 -> M=8+e
    #pragma unroll
    for (int tn = 0; tn < NBLK / 16; ++tn) {
        int col = n0 + tn * 16 + mrow;
        float bv = bias[col];
        #pragma unroll
        for (int e = 0; e < 8; ++e) {
            int row = m0 + wave * 16 + sel * 8 + e;
            float z = acc[tn][e] + bv;
            if (doSilu) z = z / (1.0f + __expf(-z));
            C[(size_t)row * Nc + col] = z;
        }
    }
}

// ---------------------------------------------------------------------------
// Low-rank products  ki[h][k][d] = sum_r a[h][k][r] * b[h][r][d]
// ---------------------------------------------------------------------------
__global__ __launch_bounds__(128)
void kind_kernel(const float* __restrict__ aP, const float* __restrict__ bP,
                 const float* __restrict__ aN, const float* __restrict__ bN,
                 float* __restrict__ kiP, float* __restrict__ kiN)
{
    const int h = blockIdx.x;
    const int d = threadIdx.x;
    const float* a = (blockIdx.y == 0) ? aP : aN;
    const float* b = (blockIdx.y == 0) ? bP : bN;
    float*      ki = (blockIdx.y == 0) ? kiP : kiN;

    #pragma unroll 1
    for (int k = 0; k < NKI; ++k) {
        float s = 0.0f;
        #pragma unroll
        for (int r = 0; r < RANKK; ++r)
            s = fmaf(a[(h * NKI + k) * RANKK + r], b[(h * RANKK + r) * HD + d], s);
        ki[(h * NKI + k) * HD + d] = s;
    }
}

// ---------------------------------------------------------------------------
// Toeplitz tap table T[h][lagIdx][d], lag = lagIdx-(n-1).
// ---------------------------------------------------------------------------
__global__ __launch_bounds__(128)
void taps_kernel(const float* __restrict__ kiP, const float* __restrict__ kiN,
                 const float* __restrict__ t_zero, float* __restrict__ T, int n)
{
    const int h = blockIdx.y;
    const int d = threadIdx.x;
    const int rows = 2 * n - 1;

    for (int rr = 0; rr < 32; ++rr) {
        int lagIdx = blockIdx.x * 32 + rr;
        if (lagIdx >= rows) return;
        int lag = lagIdx - (n - 1);
        float val;
        if (lag == 0) {
            val = t_zero[h * HD + d];
        } else {
            int l = (lag > 0) ? lag : -lag;
            const float* ki = (lag > 0) ? kiP : kiN;
            float p  = (float)(l - 1) / (float)(n - 2) * (float)(NKI - 1);
            int   lo = (int)floorf(p);
            int   hi = min(lo + 1, NKI - 1);
            float w  = p - (float)lo;
            float kv = ki[(h * NKI + lo) * HD + d] * (1.0f - w)
                     + ki[(h * NKI + hi) * HD + d] * w;
            val = kv * __expf((float)l * LN_GAMMA);   // gamma^l
        }
        T[((size_t)h * rows + lagIdx) * HD + d] = val;
    }
}

// ---------------------------------------------------------------------------
// Depthwise Toeplitz conv + gating, in-place on U:
//   G[b,i,h,d] = U[b,i,h,d] * sum_j T[h,(i-j)+(n-1),d] * V[b,j,h,d]
// ---------------------------------------------------------------------------
#define TI 32
#define TJ 32

__global__ __launch_bounds__(128)
void conv_gate_kernel(const float* __restrict__ V, const float* __restrict__ T,
                      float* __restrict__ UG, int n)
{
    __shared__ float vs[TJ * HD];                // 16 KB
    __shared__ float ts[(TI + TJ - 1) * HD];     // 31.5 KB

    const int i0 = blockIdx.x * TI;
    const int h  = blockIdx.y;
    const int bb = blockIdx.z;
    const int d  = threadIdx.x;

    float acc[TI] = {};

    const size_t rowStride = (size_t)NHEAD * HD;
    const size_t vbase = ((size_t)bb * n) * rowStride + (size_t)h * HD + d;
    const size_t tbase = (size_t)h * (2 * n - 1) * HD + d;

    for (int j0 = 0; j0 < n; j0 += TJ) {
        #pragma unroll
        for (int jj = 0; jj < TJ; ++jj)
            vs[jj * HD + d] = V[vbase + (size_t)(j0 + jj) * rowStride];
        const int lag0 = i0 - j0 + (n - 1) - (TJ - 1);
        #pragma unroll
        for (int r = 0; r < TI + TJ - 1; ++r)
            ts[r * HD + d] = T[tbase + (size_t)(lag0 + r) * HD];
        if (j0 + TJ < n)
            __builtin_prefetch(&V[vbase + (size_t)(j0 + TJ) * rowStride], 0, 0);
        __syncthreads();

        #pragma unroll 1
        for (int jj = 0; jj < TJ; ++jj) {
            float vv = vs[jj * HD + d];
            #pragma unroll
            for (int ii = 0; ii < TI; ++ii)
                acc[ii] = fmaf(ts[(ii - jj + TJ - 1) * HD + d], vv, acc[ii]);
        }
        __syncthreads();
    }

    const size_t ubase = ((size_t)bb * n + i0) * rowStride + (size_t)h * HD + d;
    #pragma unroll
    for (int ii = 0; ii < TI; ++ii) {
        size_t idx = ubase + (size_t)ii * rowStride;
        UG[idx] = UG[idx] * acc[ii];
    }
}

// ---------------------------------------------------------------------------
// Host-side orchestration
// ---------------------------------------------------------------------------
extern "C" void kernel_launch(void* const* d_in, const int* in_sizes, int n_in,
                              void* d_out, int out_size, void* d_ws, size_t ws_size,
                              hipStream_t stream)
{
    (void)in_sizes; (void)n_in; (void)out_size; (void)ws_size;

    const float* x      = (const float*)d_in[0];
    const float* Wu     = (const float*)d_in[1];
    const float* bu     = (const float*)d_in[2];
    const float* Wv     = (const float*)d_in[3];
    const float* bv     = (const float*)d_in[4];
    const float* Wo     = (const float*)d_in[5];
    const float* bo     = (const float*)d_in[6];
    const float* a_pos  = (const float*)d_in[7];
    const float* b_pos  = (const float*)d_in[8];
    const float* a_neg  = (const float*)d_in[9];
    const float* b_neg  = (const float*)d_in[10];
    const float* t_zero = (const float*)d_in[11];
    float* out = (float*)d_out;

    // ---- workspace carve-up (bytes) ----
    char* p = (char*)d_ws;
    float* U   = (float*)p;  p += (size_t)M_ROWS * D1 * 4;                 // 33.5 MB
    float* V   = (float*)p;  p += (size_t)M_ROWS * D1 * 4;                 // 33.5 MB
    float* T   = (float*)p;  p += (size_t)NHEAD * (2 * SEQ - 1) * HD * 4;  // 33.5 MB
    float* kiP = (float*)p;  p += (size_t)NHEAD * NKI * HD * 4;
    float* kiN = (float*)p;  p += (size_t)NHEAD * NKI * HD * 4;
    __bf16* Xh  = (__bf16*)p; p += (size_t)M_ROWS * DMODEL * 2;            // 8.4 MB
    __bf16* Xl  = (__bf16*)p; p += (size_t)M_ROWS * DMODEL * 2;
    __bf16* Wuh = (__bf16*)p; p += (size_t)DMODEL * D1 * 2;                // 4.2 MB each
    __bf16* Wul = (__bf16*)p; p += (size_t)DMODEL * D1 * 2;
    __bf16* Wvh = (__bf16*)p; p += (size_t)DMODEL * D1 * 2;
    __bf16* Wvl = (__bf16*)p; p += (size_t)DMODEL * D1 * 2;
    __bf16* Woh = (__bf16*)p; p += (size_t)D1 * DMODEL * 2;
    __bf16* Wol = (__bf16*)p; p += (size_t)D1 * DMODEL * 2;
    __bf16* Gh  = (__bf16*)p; p += (size_t)M_ROWS * D1 * 2;                // 16.8 MB
    __bf16* Gl  = (__bf16*)p; p += (size_t)M_ROWS * D1 * 2;

    // ---- split / transpose operands to bf16 hi/lo ----
    split_f32_bf16<<<1024, 256, 0, stream>>>(x, Xh, Xl, (size_t)M_ROWS * DMODEL);
    split_transpose_w<<<1024, 256, 0, stream>>>(Wu, Wuh, Wul, DMODEL, D1);
    split_transpose_w<<<1024, 256, 0, stream>>>(Wv, Wvh, Wvl, DMODEL, D1);
    split_transpose_w<<<1024, 256, 0, stream>>>(Wo, Woh, Wol, D1, DMODEL);

    // ---- u = silu(x@Wu+bu), v = silu(x@Wv+bv) via split-bf16 WMMA ----
    dim3 gUV(D1 / NBLK, M_ROWS / MBLK);          // (32, 32)
    gemm_bf16_split<<<gUV, 256, 0, stream>>>(Xh, Xl, Wuh, Wul, bu, U,
                                             M_ROWS, DMODEL, D1, 1);
    gemm_bf16_split<<<gUV, 256, 0, stream>>>(Xh, Xl, Wvh, Wvl, bv, V,
                                             M_ROWS, DMODEL, D1, 1);

    // ---- tap table + depthwise Toeplitz conv + gate (U -> G in place) ----
    kind_kernel<<<dim3(NHEAD, 2), 128, 0, stream>>>(a_pos, b_pos, a_neg, b_neg, kiP, kiN);
    taps_kernel<<<dim3((2 * SEQ - 1 + 31) / 32, NHEAD), 128, 0, stream>>>(kiP, kiN, t_zero, T, SEQ);
    conv_gate_kernel<<<dim3(SEQ / TI, NHEAD, BSZ), 128, 0, stream>>>(V, T, U, SEQ);

    // ---- out = G @ Wo + bo ----
    split_f32_bf16<<<1024, 256, 0, stream>>>(U, Gh, Gl, (size_t)M_ROWS * D1);
    dim3 gO(DMODEL / NBLK, M_ROWS / MBLK);       // (16, 32)
    gemm_bf16_split<<<gO, 256, 0, stream>>>(Gh, Gl, Woh, Wol, bo, out,
                                            M_ROWS, D1, DMODEL, 0);
}
